// Seq2Seq_78211354460416
// MI455X (gfx1250) — matrix operand order repaired and checked by reading, not compiled, not run on hardware
//
#include <hip/hip_runtime.h>
#include <hip/hip_bf16.h>
#include <stdint.h>

// Problem shape (fixed by the reference)
#define B_   32
#define S_   128
#define D_   4096
#define H_   1024
#define O_   4096
#define G4H_ (4 * H_)

typedef __attribute__((ext_vector_type(16))) __bf16         v16bf;
typedef __attribute__((ext_vector_type(8)))  float          v8f;
typedef __attribute__((ext_vector_type(16))) unsigned short v16u;
typedef __attribute__((ext_vector_type(8)))  unsigned short v8u;
typedef __attribute__((ext_vector_type(4)))  unsigned int   u32x4;
typedef __attribute__((ext_vector_type(8)))  int            i32x8;
typedef __attribute__((ext_vector_type(4)))  int            i32x4;

__device__ __forceinline__ unsigned short f2bf(float f) {
  // round-to-nearest-even fp32 -> bf16
  unsigned int u = __float_as_uint(f);
  u += 0x7FFFu + ((u >> 16) & 1u);
  return (unsigned short)(u >> 16);
}

// ---------------- elementwise helpers ----------------

__global__ void cvt_f32_bf16(const float* __restrict__ in,
                             unsigned short* __restrict__ out, int n) {
  int i = blockIdx.x * blockDim.x + threadIdx.x;
  if (i < n) out[i] = f2bf(in[i]);
}

__global__ void fill_zero_f32(float* __restrict__ p, int n) {
  int i = blockIdx.x * blockDim.x + threadIdx.x;
  if (i < n) p[i] = 0.f;
}

// source [B,S,D] fp32  ->  Xbf [(t*B+b), D] bf16   (time-major rows for batched GEMM)
__global__ void xt_cvt(const float* __restrict__ src,
                       unsigned short* __restrict__ dst, int total) {
  int i = blockIdx.x * blockDim.x + threadIdx.x;
  if (i >= total) return;
  int k = i % D_;
  int m = i / D_;
  int t = m / B_;
  int b = m % B_;
  dst[i] = f2bf(src[((long)b * S_ + t) * (long)D_ + k]);
}

// OUT[m*ldo + n] = b0[n] (+ b1[n])  -- seeds the split-K atomic accumulation
__global__ void bias_init(float* __restrict__ OUT, long ldo,
                          const float* __restrict__ b0,
                          const float* __restrict__ b1, int M, int N) {
  int i = blockIdx.x * blockDim.x + threadIdx.x;
  if (i >= M * N) return;
  int m = i / N, n = i % N;
  float v = b0 ? b0[n] : 0.f;
  if (b1) v += b1[n];
  OUT[(long)m * ldo + n] = v;
}

// dst[m*N+n] = bf16(src[m*lds + n])   (fc output -> decoder feedback)
__global__ void slice_cvt(const float* __restrict__ src, long lds,
                          unsigned short* __restrict__ dst, int M, int N) {
  int i = blockIdx.x * blockDim.x + threadIdx.x;
  if (i >= M * N) return;
  int m = i / N, n = i % N;
  dst[i] = f2bf(src[(long)m * lds + n]);
}

// Fused LSTM cell: gates [B,4H] fp32 (i,f,g,o), c fp32 in/out, h emitted as bf16
__global__ void lstm_cell(const float* __restrict__ gates,
                          float* __restrict__ c,
                          unsigned short* __restrict__ hbf, int BH) {
  int idx = blockIdx.x * blockDim.x + threadIdx.x;
  if (idx >= BH) return;
  int b = idx / H_, n = idx % H_;
  const float* g = gates + (long)b * G4H_;
  float gi = g[n];
  float gf = g[H_ + n];
  float gg = g[2 * H_ + n];
  float go = g[3 * H_ + n];
  float si = 1.f / (1.f + __expf(-gi));
  float sf = 1.f / (1.f + __expf(-gf));
  float so = 1.f / (1.f + __expf(-go));
  float cn = sf * c[idx] + si * tanhf(gg);
  c[idx]   = cn;
  hbf[idx] = f2bf(so * tanhf(cn));
}

// ---------------- fragment helpers ----------------

// A fragment (16x32 bf16): lane holds row m, 16 contiguous K at k+half*16
__device__ __forceinline__ v16bf load_a_frag(const unsigned short* Arow, int k,
                                             int half) {
  const v8u a0 = *(const v8u*)(Arow + k + half * 16);
  const v8u a1 = *(const v8u*)(Arow + k + half * 16 + 8);
  v16u au;
#pragma unroll
  for (int j = 0; j < 8; ++j) { au[j] = a0[j]; au[j + 8] = a1[j]; }
  return __builtin_bit_cast(v16bf, au);
}

// ---------------- Tensor Data Mover issue (D# per ISA 08_async_tensor §8) ----
// 2D tile: rows x rowlen elements of 2-byte data, row stride = stride_elems.
__device__ __forceinline__ void tdm_load_2d(unsigned lds_addr, const void* gptr,
                                            unsigned rowlen, unsigned rows,
                                            unsigned stride_elems) {
  unsigned long long ga = (unsigned long long)(uintptr_t)gptr;
  u32x4 g0;
  g0[0] = 1u;                                   // count=1, user descriptor
  g0[1] = lds_addr;                             // lds_addr (bytes)
  g0[2] = (unsigned)(ga & 0xFFFFFFFFu);         // global_addr[31:0]
  g0[3] = (unsigned)((ga >> 32) & 0x01FFFFFFu)  // global_addr[56:32]
          | (2u << 30);                         // type=2 ("image")
  i32x8 g1;
  g1[0] = (int)(1u << 16);                      // data_size=1 (2 bytes)
  g1[1] = (int)(rowlen << 16);                  // tensor_dim0[15:0] @ [63:48]
  g1[2] = (int)((rowlen >> 16) | (rows << 16)); // tensor_dim0 hi | tensor_dim1 lo
  g1[3] = (int)(rowlen << 16);                  // tensor_dim1 hi(0) | tile_dim0
  g1[4] = (int)rows;                            // tile_dim1 | tile_dim2(0)
  g1[5] = (int)stride_elems;                    // tensor_dim0_stride[31:0]
  g1[6] = 0;                                    // stride hi | dim1_stride lo
  g1[7] = 0;                                    // dim1_stride hi
  const i32x4 z4 = {0, 0, 0, 0};
#if __clang_major__ >= 23
  const i32x8 z8 = {0, 0, 0, 0, 0, 0, 0, 0};
  __builtin_amdgcn_tensor_load_to_lds(g0, g1, z4, z4, z8, 0);
#else
  __builtin_amdgcn_tensor_load_to_lds(g0, g1, z4, z4, 0);
#endif
}

// ---------------- batched GEMM: TDM-staged B, 64x64 tile per block ----------
// OUT[M,N] = A[M,K](bf16) @ W[N,K]^T(bf16) + bias0+bias1. 4 waves/block, each
// wave one 16-row m-tile x 64 cols; B tile (64 rows x 32 k) DMA'd to LDS and
// double-buffered with TENSORcnt.
__global__ void gemm_wmma_tdm(const unsigned short* __restrict__ A, long lda,
                              const unsigned short* __restrict__ W, long ldw,
                              int M, int N, int K,
                              float* __restrict__ OUT, long ldo,
                              const float* __restrict__ bias0,
                              const float* __restrict__ bias1) {
  __shared__ unsigned short Bt[2][64 * 32];  // 2 x 4 KB

  const int lane = threadIdx.x & 31;
  const int wave = threadIdx.x >> 5;
  const int half = lane >> 4;
  const int lm   = lane & 15;

  const int mblocks = M >> 6;
  const int bm = blockIdx.x % mblocks;
  const int bn = blockIdx.x / mblocks;
  const int m0 = bm * 64 + wave * 16;
  const int n0 = bn * 64;

  v8f acc[4];
#pragma unroll
  for (int nt = 0; nt < 4; ++nt) {
    const float cv = bias0[n0 + nt * 16 + lm] + bias1[n0 + nt * 16 + lm];
#pragma unroll
    for (int r = 0; r < 8; ++r) acc[nt][r] = cv;
  }

  const unsigned short* Arow  = A + (long)(m0 + lm) * lda;
  const unsigned short* Wtile = W + (long)n0 * ldw;  // rows n0..n0+63

  if (wave == 0)
    tdm_load_2d((unsigned)(uintptr_t)&Bt[0][0], Wtile, 32, 64, (unsigned)ldw);

  int buf = 0;
  for (int k = 0; k < K; k += 32) {
    if (wave == 0) {
      if (k + 32 < K) {
        tdm_load_2d((unsigned)(uintptr_t)&Bt[buf ^ 1][0], Wtile + (k + 32), 32,
                    64, (unsigned)ldw);
        __builtin_amdgcn_s_wait_tensorcnt(1);  // previous tile landed
      } else {
        __builtin_amdgcn_s_wait_tensorcnt(0);
      }
    }
    __syncthreads();  // publish LDS tile to all 4 waves

    const v16bf av = load_a_frag(Arow, k, half);
#pragma unroll
    for (int nt = 0; nt < 4; ++nt) {
      // B fragment (32x16): lane col n, 16 contiguous K at k+half*16 -> ds_load_b128 x2
      const v16u bu =
          *(const v16u*)(&Bt[buf][(nt * 16 + lm) * 32 + half * 16]);
      const v16bf bv = __builtin_bit_cast(v16bf, bu);
      acc[nt] = __builtin_amdgcn_wmma_f32_16x16x32_bf16(
          false, av, false, bv, (short)0, acc[nt], false, false);
    }
    __syncthreads();  // tile consumed; safe to overwrite next round
    buf ^= 1;
  }

#pragma unroll
  for (int nt = 0; nt < 4; ++nt) {
    const int n = n0 + nt * 16 + lm;
#pragma unroll
    for (int r = 0; r < 8; ++r)
      OUT[(long)(m0 + half * 8 + r) * ldo + n] = acc[nt][r];
  }
}

// ---------------- per-step GEMM: direct global, split-K, atomic epilogue ----
// OUT[M,N] += A[M,K](bf16) @ W[N,K]^T(bf16). OUT pre-seeded with biases.
__global__ void gemm_step_wmma(const unsigned short* __restrict__ A, long lda,
                               const unsigned short* __restrict__ W, long ldw,
                               int M, int N, int K, int splitK,
                               float* __restrict__ OUT, long ldo) {
  const int lane = threadIdx.x & 31;
  const int wave = threadIdx.x >> 5;
  const int mt = M >> 4;
  const int ng = N >> 4;
  int id = blockIdx.x * (blockDim.x >> 5) + wave;
  if (id >= mt * ng * splitK) return;
  const int ks = id % splitK; id /= splitK;
  const int im = id % mt;
  const int in = id / mt;
  const int m0 = im * 16;
  const int n0 = in * 16;
  const int Kc = K / splitK;
  const int k0 = ks * Kc;

  const int half = lane >> 4;
  const int lm   = lane & 15;

  v8f acc;
#pragma unroll
  for (int r = 0; r < 8; ++r) acc[r] = 0.f;

  const unsigned short* Arow = A + (long)(m0 + lm) * lda;
  const unsigned short* Wrow = W + (long)(n0 + lm) * ldw + half * 16;

  for (int k = k0; k < k0 + Kc; k += 32) {
    const v16bf av = load_a_frag(Arow, k, half);
    const v16u bu = *(const v16u*)(Wrow + k);
    const v16bf bv = __builtin_bit_cast(v16bf, bu);
    acc = __builtin_amdgcn_wmma_f32_16x16x32_bf16(false, av, false, bv,
                                                  (short)0, acc, false, false);
  }

  const int n = n0 + lm;
#pragma unroll
  for (int r = 0; r < 8; ++r)
    unsafeAtomicAdd(&OUT[(long)(m0 + half * 8 + r) * ldo + n],
                    acc[r]);  // global_atomic_add_f32 (L2)
}

// ---------------- host side ----------------

static inline int gcdiv(int a, int b) { return (a + b - 1) / b; }

static inline void launch_big(const unsigned short* A, long lda,
                              const unsigned short* W, long ldw, int M, int N,
                              int K, float* OUT, long ldo, const float* b0,
                              const float* b1, hipStream_t s) {
  int blocks = (M / 64) * (N / 64);
  gemm_wmma_tdm<<<blocks, 128, 0, s>>>(A, lda, W, ldw, M, N, K, OUT, ldo, b0, b1);
}

static inline void launch_step(const unsigned short* A, long lda,
                               const unsigned short* W, long ldw, int M, int N,
                               int K, int splitK, float* OUT, long ldo,
                               hipStream_t s) {
  int tiles = (M / 16) * (N / 16) * splitK;
  gemm_step_wmma<<<(tiles + 3) / 4, 128, 0, s>>>(A, lda, W, ldw, M, N, K,
                                                 splitK, OUT, ldo);
}

extern "C" void kernel_launch(void* const* d_in, const int* in_sizes, int n_in,
                              void* d_out, int out_size, void* d_ws,
                              size_t ws_size, hipStream_t stream) {
  (void)in_sizes; (void)n_in; (void)out_size; (void)ws_size;

  const float* source = (const float*)d_in[0];
  const float* e_wih0 = (const float*)d_in[1];
  const float* e_whh0 = (const float*)d_in[2];
  const float* e_bih0 = (const float*)d_in[3];
  const float* e_bhh0 = (const float*)d_in[4];
  const float* e_wih1 = (const float*)d_in[5];
  const float* e_whh1 = (const float*)d_in[6];
  const float* e_bih1 = (const float*)d_in[7];
  const float* e_bhh1 = (const float*)d_in[8];
  const float* d_wih0 = (const float*)d_in[9];
  const float* d_whh0 = (const float*)d_in[10];
  const float* d_bih0 = (const float*)d_in[11];
  const float* d_bhh0 = (const float*)d_in[12];
  const float* d_wih1 = (const float*)d_in[13];
  const float* d_whh1 = (const float*)d_in[14];
  const float* d_bih1 = (const float*)d_in[15];
  const float* d_bhh1 = (const float*)d_in[16];
  const float* fc_w   = (const float*)d_in[17];
  const float* fc_b   = (const float*)d_in[18];
  float* out = (float*)d_out;  // [B, S, O]

  // ---- workspace sub-allocation (256-aligned); total ~236 MB ----
  char* ws = (char*)d_ws;
  size_t off = 0;
  auto alloc = [&](size_t bytes) -> char* {
    char* p = ws + off;
    off += (bytes + 255) & ~(size_t)255;
    return p;
  };
  unsigned short* wb_ewih0 = (unsigned short*)alloc((size_t)G4H_ * D_ * 2);
  unsigned short* wb_ewhh0 = (unsigned short*)alloc((size_t)G4H_ * H_ * 2);
  unsigned short* wb_ewih1 = (unsigned short*)alloc((size_t)G4H_ * H_ * 2);
  unsigned short* wb_ewhh1 = (unsigned short*)alloc((size_t)G4H_ * H_ * 2);
  unsigned short* wb_dwih0 = (unsigned short*)alloc((size_t)G4H_ * D_ * 2);
  unsigned short* wb_dwhh0 = (unsigned short*)alloc((size_t)G4H_ * H_ * 2);
  unsigned short* wb_dwih1 = (unsigned short*)alloc((size_t)G4H_ * H_ * 2);
  unsigned short* wb_dwhh1 = (unsigned short*)alloc((size_t)G4H_ * H_ * 2);
  unsigned short* wb_fcw   = (unsigned short*)alloc((size_t)O_ * H_ * 2);
  unsigned short* Xbf      = (unsigned short*)alloc((size_t)S_ * B_ * D_ * 2);
  float*          G        = (float*)alloc((size_t)S_ * B_ * G4H_ * 4);  // reused for both encoder layers
  unsigned short* ys0bf    = (unsigned short*)alloc((size_t)S_ * B_ * H_ * 2);
  unsigned short* h1bf     = (unsigned short*)alloc((size_t)B_ * H_ * 2);
  unsigned short* dh0bf    = (unsigned short*)alloc((size_t)B_ * H_ * 2);
  unsigned short* predbf   = (unsigned short*)alloc((size_t)B_ * O_ * 2);
  float*          c0       = (float*)alloc((size_t)B_ * H_ * 4);
  float*          c1       = (float*)alloc((size_t)B_ * H_ * 4);
  float*          dgates   = (float*)alloc((size_t)B_ * G4H_ * 4);

  const int BH = B_ * H_;
  const int SB = S_ * B_;
  const long ldout = (long)S_ * O_;

  // ---- one-time: weights fp32 -> bf16 (L2-resident thereafter), state zeroing ----
  struct { const float* s; unsigned short* d; int n; } cv[9] = {
      {e_wih0, wb_ewih0, G4H_ * D_}, {e_whh0, wb_ewhh0, G4H_ * H_},
      {e_wih1, wb_ewih1, G4H_ * H_}, {e_whh1, wb_ewhh1, G4H_ * H_},
      {d_wih0, wb_dwih0, G4H_ * D_}, {d_whh0, wb_dwhh0, G4H_ * H_},
      {d_wih1, wb_dwih1, G4H_ * H_}, {d_whh1, wb_dwhh1, G4H_ * H_},
      {fc_w,   wb_fcw,   O_ * H_}};
  for (int i = 0; i < 9; ++i)
    cvt_f32_bf16<<<gcdiv(cv[i].n, 256), 256, 0, stream>>>(cv[i].s, cv[i].d,
                                                          cv[i].n);
  fill_zero_f32<<<gcdiv(BH, 256), 256, 0, stream>>>(c0, BH);
  fill_zero_f32<<<gcdiv(BH, 256), 256, 0, stream>>>(c1, BH);
  xt_cvt<<<gcdiv(SB * D_, 256), 256, 0, stream>>>(source, Xbf, SB * D_);

  // ---- encoder layer 0: batched input projection (TDM-staged WMMA GEMM) ----
  launch_big(Xbf, D_, wb_ewih0, D_, SB, G4H_, D_, G, G4H_, e_bih0, e_bhh0,
             stream);
  for (int t = 0; t < S_; ++t) {
    if (t > 0)  // gates_t += h_{t-1} @ Whh0^T   (split-K atomics)
      launch_step(ys0bf + (size_t)(t - 1) * BH, H_, wb_ewhh0, H_, B_, G4H_, H_,
                  4, G + (size_t)t * B_ * G4H_, G4H_, stream);
    lstm_cell<<<gcdiv(BH, 256), 256, 0, stream>>>(
        G + (size_t)t * B_ * G4H_, c0, ys0bf + (size_t)t * BH, BH);
  }

  // ---- encoder layer 1: batched projection of ys0 (reuse G) ----
  launch_big(ys0bf, H_, wb_ewih1, H_, SB, G4H_, H_, G, G4H_, e_bih1, e_bhh1,
             stream);
  for (int t = 0; t < S_; ++t) {
    if (t > 0)
      launch_step(h1bf, H_, wb_ewhh1, H_, B_, G4H_, H_, 4,
                  G + (size_t)t * B_ * G4H_, G4H_, stream);
    lstm_cell<<<gcdiv(BH, 256), 256, 0, stream>>>(
        G + (size_t)t * B_ * G4H_, c1, h1bf, BH);
  }
  // decoder inherits (h0,c0) = encoder L0 final (ys0bf last slice, c0),
  // (h1,c1) = encoder L1 final (h1bf, c1).

  // ---- autoregressive decoder ----
  for (int t = 0; t < S_; ++t) {
    // layer 0: gates = x@Wih0^T + h0@Whh0^T + biases  (x0 = 0 -> skip x GEMM)
    bias_init<<<gcdiv(B_ * G4H_, 256), 256, 0, stream>>>(dgates, G4H_, d_bih0,
                                                         d_bhh0, B_, G4H_);
    if (t > 0)
      launch_step(predbf, O_, wb_dwih0, D_, B_, G4H_, D_, 8, dgates, G4H_,
                  stream);
    const unsigned short* h0src =
        (t == 0) ? ys0bf + (size_t)(S_ - 1) * BH : dh0bf;
    launch_step(h0src, H_, wb_dwhh0, H_, B_, G4H_, H_, 4, dgates, G4H_, stream);
    lstm_cell<<<gcdiv(BH, 256), 256, 0, stream>>>(dgates, c0, dh0bf, BH);

    // layer 1: gates = h0@Wih1^T + h1@Whh1^T + biases
    bias_init<<<gcdiv(B_ * G4H_, 256), 256, 0, stream>>>(dgates, G4H_, d_bih1,
                                                         d_bhh1, B_, G4H_);
    launch_step(dh0bf, H_, wb_dwih1, H_, B_, G4H_, H_, 4, dgates, G4H_, stream);
    launch_step(h1bf, H_, wb_dwhh1, H_, B_, G4H_, H_, 4, dgates, G4H_, stream);
    lstm_cell<<<gcdiv(BH, 256), 256, 0, stream>>>(dgates, c1, h1bf, BH);

    // fc: pred = h1 @ fc_w^T + fc_b  -> d_out[:, t, :], plus bf16 feedback
    bias_init<<<gcdiv(B_ * O_, 256), 256, 0, stream>>>(out + (size_t)t * O_,
                                                       ldout, fc_b, nullptr,
                                                       B_, O_);
    launch_step(h1bf, H_, wb_fcw, H_, B_, O_, H_, 4, out + (size_t)t * O_,
                ldout, stream);
    slice_cvt<<<gcdiv(B_ * O_, 256), 256, 0, stream>>>(
        out + (size_t)t * O_, ldout, predbf, B_, O_);
  }
}
